// MENet_single_88665304858931
// MI455X (gfx1250) — matrix operand order repaired and checked by reading, not compiled, not run on hardware
//
#include <hip/hip_runtime.h>

typedef __attribute__((ext_vector_type(16))) _Float16 v16h;
typedef __attribute__((ext_vector_type(8)))  _Float16 v8h;
typedef __attribute__((ext_vector_type(8)))  float    v8f;

#define BN_SCALE 0.99999500003749968f

// ---------------------------------------------------------------------------
// Elementwise / gather / sampling kernels (wave32 VALU path)
// ---------------------------------------------------------------------------

__global__ void k_transpose_xyz(const float* __restrict__ in, float* __restrict__ out,
                                int B, int N) {
  long long t = (long long)blockIdx.x * blockDim.x + threadIdx.x;
  long long tot = (long long)B * N * 3;
  if (t >= tot) return;
  int d = (int)(t % 3);
  int n = (int)((t / 3) % N);
  int b = (int)(t / (3LL * N));
  out[t] = in[(size_t)b * 3 * N + (size_t)d * N + n];
}

// Farthest point sampling: one block per batch, dist cached in LDS.
__global__ void k_fps(const float* __restrict__ pts, int N, int npoint,
                      int* __restrict__ idxOut) {
  __shared__ float dist[4096];
  __shared__ float redV[256];
  __shared__ int   redI[256];
  __shared__ int   curFar;
  const int b = blockIdx.x;
  const float* P = pts + (size_t)b * N * 3;
  for (int i = threadIdx.x; i < N; i += blockDim.x) dist[i] = 1e10f;
  if (threadIdx.x == 0) curFar = 0;
  __syncthreads();
  for (int it = 0; it < npoint; ++it) {
    int far = curFar;
    if (threadIdx.x == 0) idxOut[(size_t)b * npoint + it] = far;
    float fx = P[far * 3 + 0], fy = P[far * 3 + 1], fz = P[far * 3 + 2];
    float bestV = -1.0f; int bestI = 0;
    for (int i = threadIdx.x; i < N; i += blockDim.x) {
      float dx = P[i * 3 + 0] - fx;
      float dy = P[i * 3 + 1] - fy;
      float dz = P[i * 3 + 2] - fz;
      float d = dx * dx + dy * dy + dz * dz;
      float nd = fminf(dist[i], d);
      dist[i] = nd;
      if (nd > bestV) { bestV = nd; bestI = i; }
    }
    redV[threadIdx.x] = bestV; redI[threadIdx.x] = bestI;
    __syncthreads();
    for (int s = 128; s > 0; s >>= 1) {
      if (threadIdx.x < s) {
        float ov = redV[threadIdx.x + s]; int oi = redI[threadIdx.x + s];
        if (ov > redV[threadIdx.x] ||
            (ov == redV[threadIdx.x] && oi < redI[threadIdx.x])) {
          redV[threadIdx.x] = ov; redI[threadIdx.x] = oi;
        }
      }
      __syncthreads();
    }
    if (threadIdx.x == 0) curFar = redI[0];
    __syncthreads();
  }
}

__global__ void k_gather3(const float* __restrict__ pts, const int* __restrict__ idx,
                          float* __restrict__ out, int N, int S, int B) {
  int t = blockIdx.x * blockDim.x + threadIdx.x;
  int tot = B * S * 3;
  if (t >= tot) return;
  int d = t % 3;
  int s = (t / 3) % S;
  int b = t / (3 * S);
  int j = idx[b * S + s];
  out[t] = pts[((size_t)b * N + j) * 3 + d];
}

// First-K-ascending in-radius indices, padded with the first hit (== reference).
__global__ void k_ball_query(const float* __restrict__ xyz, const float* __restrict__ nxyz,
                             float r2, int K, int N, int S, int G, int* __restrict__ out) {
  int t = blockIdx.x * blockDim.x + threadIdx.x;
  if (t >= G) return;
  int b = t / S;
  const float* X = xyz + (size_t)b * N * 3;
  float cx = nxyz[(size_t)t * 3 + 0];
  float cy = nxyz[(size_t)t * 3 + 1];
  float cz = nxyz[(size_t)t * 3 + 2];
  int* o = out + (size_t)t * K;
  int cnt = 0, first = -1;
  for (int j = 0; j < N && cnt < K; ++j) {
    float dx = X[j * 3 + 0] - cx;
    float dy = X[j * 3 + 1] - cy;
    float dz = X[j * 3 + 2] - cz;
    if (dx * dx + dy * dy + dz * dz <= r2) {
      if (first < 0) first = j;
      o[cnt++] = j;
    }
  }
  if (first < 0) first = 0;
  for (; cnt < K; ++cnt) o[cnt] = first;
}

// Build grouped features [gathered point feats (Cp) | grouped xyz-center (3) | zero pad]
// as fp16 rows ready for the WMMA GEMM. Processes groups [g0, g0+GC).
__global__ void k_build_group_feat(const _Float16* __restrict__ ptsH, int Cp,
                                   const float* __restrict__ xyz,
                                   const float* __restrict__ nxyz,
                                   const int* __restrict__ idx,
                                   int K, int g0, int GC, int N, int S,
                                   _Float16* __restrict__ A, int cinPad) {
  long long t = (long long)blockIdx.x * blockDim.x + threadIdx.x;
  long long tot = (long long)GC * K * cinPad;
  if (t >= tot) return;
  int c = (int)(t % cinPad);
  long long rk = t / cinPad;
  int k = (int)(rk % K);
  int gi = (int)(rk / K);
  int g = g0 + gi;
  int b = g / S;
  int j = idx[(size_t)g * K + k];
  _Float16 v = (_Float16)0.0f;
  if (c < Cp) {
    v = ptsH[((size_t)b * N + j) * Cp + c];
  } else if (c < Cp + 3) {
    int d = c - Cp;
    v = (_Float16)(xyz[((size_t)b * N + j) * 3 + d] - nxyz[(size_t)g * 3 + d]);
  }
  A[t] = v;
}

// Max over the K group members; writes a channel slice of the stage output.
__global__ void k_reduce_max(const _Float16* __restrict__ in, int K, int C,
                             int g0, int GC, void* __restrict__ out, int ldo,
                             int cOff, int outF16) {
  long long t = (long long)blockIdx.x * blockDim.x + threadIdx.x;
  long long tot = (long long)GC * C;
  if (t >= tot) return;
  int c = (int)(t % C);
  int gi = (int)(t / C);
  const _Float16* p = in + (size_t)gi * K * C + c;
  float m = -3.4e38f;
  for (int k = 0; k < K; ++k) m = fmaxf(m, (float)p[(size_t)k * C]);
  size_t oidx = (size_t)(g0 + gi) * ldo + cOff + c;
  if (outF16) ((_Float16*)out)[oidx] = (_Float16)m;
  else        ((float*)out)[oidx] = m;
}

// SA3 input: [l2_xyz(3) | radial(1) | l2_points(640) | pad→672] as fp16.
__global__ void k_build_feat3(const float* __restrict__ l2xyz,
                              const _Float16* __restrict__ l2pts,
                              _Float16* __restrict__ A, int R, int cinPad) {
  long long t = (long long)blockIdx.x * blockDim.x + threadIdx.x;
  long long tot = (long long)R * cinPad;
  if (t >= tot) return;
  int c = (int)(t % cinPad);
  int r = (int)(t / cinPad);
  _Float16 v = (_Float16)0.0f;
  if (c < 3) {
    v = (_Float16)l2xyz[(size_t)r * 3 + c];
  } else if (c == 3) {
    float x = l2xyz[(size_t)r * 3 + 0];
    float y = l2xyz[(size_t)r * 3 + 1];
    float z = l2xyz[(size_t)r * 3 + 2];
    v = (_Float16)sqrtf(x * x + y * y + z * z);
  } else if (c < 644) {
    v = l2pts[(size_t)r * 640 + (c - 4)];
  }
  A[t] = v;
}

__global__ void k_l2norm_rows(const float* __restrict__ in, _Float16* __restrict__ out,
                              int R, int C, int Cpad) {
  int r = blockIdx.x * blockDim.x + threadIdx.x;
  if (r >= R) return;
  const float* x = in + (size_t)r * C;
  float ss = 0.0f;
  for (int c = 0; c < C; ++c) ss += x[c] * x[c];
  float nrm = sqrtf(ss);
  nrm = nrm > 1e-12f ? nrm : 1e-12f;
  float inv = 1.0f / nrm;
  _Float16* o = out + (size_t)r * Cpad;
  for (int c = 0; c < Cpad; ++c)
    o[c] = (c < C) ? (_Float16)(x[c] * inv) : (_Float16)0.0f;
}

// memory_w (16,64) fp32 -> transposed fp16 (64,32) zero-padded (WMMA B operand for addr@M).
__global__ void k_memw_T(const float* __restrict__ w, _Float16* __restrict__ out) {
  int t = blockIdx.x * blockDim.x + threadIdx.x;
  if (t >= 64 * 32) return;
  int k = t % 32, n = t / 32;
  out[t] = (k < 16) ? (_Float16)w[k * 64 + n] : (_Float16)0.0f;
}

__global__ void k_softmax16(const float* __restrict__ s, _Float16* __restrict__ out, int R) {
  int r = blockIdx.x * blockDim.x + threadIdx.x;
  if (r >= R) return;
  const float* x = s + (size_t)r * 16;
  float m = x[0];
  for (int c = 1; c < 16; ++c) m = fmaxf(m, x[c]);
  float e[16];
  float sum = 0.0f;
  for (int c = 0; c < 16; ++c) { e[c] = expf(x[c] - m); sum += e[c]; }
  float inv = 1.0f / sum;
  _Float16* o = out + (size_t)r * 32;
  for (int c = 0; c < 16; ++c) o[c] = (_Float16)(e[c] * inv);
  for (int c = 16; c < 32; ++c) o[c] = (_Float16)0.0f;
}

// fp32 weights (cout,cin) -> fp16 padded (coutP, cinP).
__global__ void k_cvt_w(const float* __restrict__ w, _Float16* __restrict__ out,
                        int cout, int cin, int coutP, int cinP) {
  long long t = (long long)blockIdx.x * blockDim.x + threadIdx.x;
  long long tot = (long long)coutP * cinP;
  if (t >= tot) return;
  int c = (int)(t % cinP);
  int r = (int)(t / cinP);
  out[t] = (r < cout && c < cin) ? (_Float16)w[(size_t)r * cin + c] : (_Float16)0.0f;
}

__global__ void k_logsoftmax(const float* __restrict__ logits, float* __restrict__ out,
                             int R, int C) {
  int r = blockIdx.x * blockDim.x + threadIdx.x;
  if (r >= R) return;
  const float* x = logits + (size_t)r * C;
  float m = -3.4e38f;
  for (int c = 0; c < C; ++c) m = fmaxf(m, x[c]);
  float s = 0.0f;
  for (int c = 0; c < C; ++c) s += expf(x[c] - m);
  float ls = logf(s);
  for (int c = 0; c < C; ++c) out[(size_t)r * C + c] = x[c] - m - ls;
}

// ---------------------------------------------------------------------------
// WMMA GEMM: Out[M,Nout] = act(A[M,Kdim] * W[NoutP,Kdim]^T + bias)
// 8 waves/block; each wave owns a 16x32 tile (2 accumulators sharing one A
// fragment). Weight panel (32 rows x 128 K halves) staged in LDS once per
// 128-K chunk -> one barrier pair per 4 WMMA steps instead of per step.
// act: 0 -> x (+bias), 1 -> relu(BN_SCALE*(x+bias)). outF16: fp16 vs fp32 store.
// ---------------------------------------------------------------------------
__global__ void __launch_bounds__(256)
k_wmma_gemm(const _Float16* __restrict__ A, int lda,
            const _Float16* __restrict__ W, int ldw,
            const float* __restrict__ bias,
            void* __restrict__ Out, int ldo,
            int M, int Nout, int Kdim, int act, int outF16) {
  __shared__ __align__(32) _Float16 Bs[32 * 128];   // 8 KB weight panel
  const int lane = threadIdx.x & 31;
  const int wave = threadIdx.x >> 5;
  const int rowBase = blockIdx.x * 128 + wave * 16;
  const int colBase = blockIdx.y * 32;
  const int mL = lane & 15;
  const int hiHalf = (lane >= 16) ? 1 : 0;
  int aRow = rowBase + mL;
  if (aRow >= M) aRow = M - 1;                 // clamp; stores are guarded below
  const _Float16* Arow = A + (size_t)aRow * lda;
  const _Float16* Wtile = W + (size_t)colBase * ldw;
  const int aOff = hiHalf ? 16 : 0;            // A frag: K = e + 16*hiHalf
  const int bOff = hiHalf ? 16 : 0;            // B frag: K = e + 16*hiHalf

  v8f acc0 = {};
  v8f acc1 = {};
  for (int k0 = 0; k0 < Kdim; k0 += 128) {
    int kChunk = Kdim - k0;
    if (kChunk > 128) kChunk = 128;
    __syncthreads();
    {
      int perRow = kChunk >> 3;                // 16-byte units per row (4..16)
      int total = 32 * perRow;
      for (int t = threadIdx.x; t < total; t += 256) {
        int r = t / perRow;
        int u = t - r * perRow;
        *(uint4*)(&Bs[r * 128 + u * 8]) =
            *(const uint4*)(Wtile + (size_t)r * ldw + k0 + u * 8);
      }
    }
    __syncthreads();
#if defined(__gfx1250__)
    if (k0 + 128 < Kdim) __builtin_prefetch(Arow + k0 + 128, 0, 1);
#endif
    for (int ks = 0; ks < kChunk; ks += 32) {
      v16h a;
      {
        v8h lo = *(const v8h*)(Arow + k0 + ks + aOff);
        v8h hi = *(const v8h*)(Arow + k0 + ks + 8 + aOff);
#pragma unroll
        for (int i = 0; i < 8; ++i) { a[i] = lo[i]; a[i + 8] = hi[i]; }
      }
      v16h b0 = *(const v16h*)(&Bs[mL * 128 + ks + bOff]);
      v16h b1 = *(const v16h*)(&Bs[(16 + mL) * 128 + ks + bOff]);
      acc0 = __builtin_amdgcn_wmma_f32_16x16x32_f16(false, a, false, b0,
                                                    (short)0, acc0, false, false);
      acc1 = __builtin_amdgcn_wmma_f32_16x16x32_f16(false, a, false, b1,
                                                    (short)0, acc1, false, false);
    }
  }

  const int rowOff = hiHalf ? 8 : 0;
  {
    int col = colBase + mL;
    float bv = (bias != nullptr && col < Nout) ? bias[col] : 0.0f;
#pragma unroll
    for (int r = 0; r < 8; ++r) {
      int row = rowBase + r + rowOff;
      if (row < M && col < Nout) {
        float v = acc0[r] + bv;
        if (act) { v *= BN_SCALE; v = v > 0.0f ? v : 0.0f; }
        if (outF16) ((_Float16*)Out)[(size_t)row * ldo + col] = (_Float16)v;
        else        ((float*)Out)[(size_t)row * ldo + col] = v;
      }
    }
  }
  {
    int col = colBase + 16 + mL;
    float bv = (bias != nullptr && col < Nout) ? bias[col] : 0.0f;
#pragma unroll
    for (int r = 0; r < 8; ++r) {
      int row = rowBase + r + rowOff;
      if (row < M && col < Nout) {
        float v = acc1[r] + bv;
        if (act) { v *= BN_SCALE; v = v > 0.0f ? v : 0.0f; }
        if (outF16) ((_Float16*)Out)[(size_t)row * ldo + col] = (_Float16)v;
        else        ((float*)Out)[(size_t)row * ldo + col] = v;
      }
    }
  }
}

// ---------------------------------------------------------------------------
// Host orchestration
// ---------------------------------------------------------------------------
extern "C" void kernel_launch(void* const* d_in, const int* in_sizes, int n_in,
                              void* d_out, int out_size, void* d_ws, size_t ws_size,
                              hipStream_t stream) {
  (void)in_sizes; (void)n_in; (void)out_size; (void)ws_size;
  const int B = 8, N0 = 4096;
  auto pad32 = [](int x) { return (x + 31) & ~31; };

  // deterministic bump allocator over workspace
  char* base = (char*)d_ws;
  size_t cur = 0;
  auto alloc = [&](size_t bytes) -> void* {
    void* p = base + cur;
    cur = (cur + bytes + 255) & ~(size_t)255;
    return p;
  };

  _Float16* PING   = (_Float16*)alloc((size_t)25 << 20);
  _Float16* PONG   = (_Float16*)alloc((size_t)25 << 20);
  float* xyzT      = (float*)alloc((size_t)B * N0 * 3 * 4);
  int*   ballIdx   = (int*)alloc((size_t)B * 512 * 128 * 4);
  int*   fpsIdx1   = (int*)alloc((size_t)B * 512 * 4);
  float* l1xyz     = (float*)alloc((size_t)B * 512 * 3 * 4);
  _Float16* l1pts  = (_Float16*)alloc((size_t)B * 512 * 320 * 2);
  int*   fpsIdx2   = (int*)alloc((size_t)B * 128 * 4);
  float* l2xyz     = (float*)alloc((size_t)B * 128 * 3 * 4);
  _Float16* l2pts  = (_Float16*)alloc((size_t)B * 128 * 640 * 2);
  _Float16* l3pts  = (_Float16*)alloc((size_t)B * 128 * 1024 * 2);
  int*   fpsIdx3   = (int*)alloc((size_t)B * 32 * 4);
  float* appxyz    = (float*)alloc((size_t)B * 32 * 3 * 4);
  float* memfeat   = (float*)alloc((size_t)256 * 64 * 4);
  _Float16* q_h    = (_Float16*)alloc((size_t)256 * 64 * 2);
  _Float16* memn_h = (_Float16*)alloc((size_t)32 * 64 * 2);  // 32 rows: LDS panel reads rows 0..31
  _Float16* memT_h = (_Float16*)alloc((size_t)64 * 32 * 2);
  float* s_f       = (float*)alloc((size_t)256 * 16 * 4);
  _Float16* addr_h = (_Float16*)alloc((size_t)256 * 32 * 2);
  _Float16* mf_h   = (_Float16*)alloc((size_t)256 * 64 * 2);
  _Float16* mm1_h  = (_Float16*)alloc((size_t)256 * 128 * 2);
  _Float16* mm2_h  = (_Float16*)alloc((size_t)256 * 256 * 2);
  _Float16* x_h    = (_Float16*)alloc((size_t)B * 1280 * 2);
  _Float16* t1_h   = (_Float16*)alloc((size_t)B * 512 * 2);
  _Float16* t2_h   = (_Float16*)alloc((size_t)B * 256 * 2);
  float* logits    = (float*)alloc((size_t)B * 40 * 4);

  struct WH { const _Float16* w; const float* b; int cout; int cin; };
  auto cvtW = [&](int wIdx, int cout, int cin) -> WH {
    int cp = pad32(cout), kp = pad32(cin);   // rows padded to 32 for the 32-wide N tile
    _Float16* wh = (_Float16*)alloc((size_t)cp * kp * 2);
    long long tot = (long long)cp * kp;
    k_cvt_w<<<(unsigned)((tot + 255) / 256), 256, 0, stream>>>(
        (const float*)d_in[wIdx], wh, cout, cin, cp, kp);
    return WH{wh, (const float*)d_in[wIdx + 1], cout, cin};
  };

  WH sa1w[3][3] = {
    { cvtW(1, 32, 3),   cvtW(3, 32, 32),    cvtW(5, 64, 32) },
    { cvtW(7, 64, 3),   cvtW(9, 64, 64),    cvtW(11, 128, 64) },
    { cvtW(13, 64, 3),  cvtW(15, 96, 64),   cvtW(17, 128, 96) } };
  WH sa2w[3][3] = {
    { cvtW(19, 64, 323),  cvtW(21, 64, 64),   cvtW(23, 128, 64) },
    { cvtW(25, 128, 323), cvtW(27, 128, 128), cvtW(29, 256, 128) },
    { cvtW(31, 128, 323), cvtW(33, 128, 128), cvtW(35, 256, 128) } };
  WH sa3w[3] = { cvtW(37, 256, 644), cvtW(39, 512, 256), cvtW(41, 1024, 512) };
  WH appw[3] = { cvtW(43, 512, 1027), cvtW(45, 256, 512), cvtW(47, 64, 256) };
  WH memw[2] = { cvtW(49, 128, 64), cvtW(51, 256, 128) };
  WH fcw[3]  = { cvtW(54, 512, 1280), cvtW(56, 256, 512), cvtW(58, 40, 256) };

  auto gemm = [&](const _Float16* A, const _Float16* Wh, const float* bias,
                  void* out, int ldo, int M, int Nout, int Kpad, int act, int of16) {
    dim3 g((unsigned)((M + 127) / 128), (unsigned)((Nout + 31) / 32));
    k_wmma_gemm<<<g, 256, 0, stream>>>(A, Kpad, Wh, Kpad, bias, out, ldo,
                                       M, Nout, Kpad, act, of16);
  };
  auto g3 = [&](const _Float16* A, const WH& w, void* out, int ldo, int M,
                int act, int of16) {
    gemm(A, w.w, w.b, out, ldo, M, w.cout, pad32(w.cin), act, of16);
  };

  const int ROWS_CAP = 32768;
  auto runScale = [&](const float* xyzSrc, int N, const float* nxyz, int S,
                      const _Float16* ptsH, int Cp, float radius, int K,
                      const WH* L, void* outPts, int ldOut, int cOff, int outF16) {
    int G = B * S;
    int cinPad = pad32(Cp + 3);
    k_ball_query<<<(G + 255) / 256, 256, 0, stream>>>(xyzSrc, nxyz, radius * radius,
                                                      K, N, S, G, ballIdx);
    int GC = ROWS_CAP / K;
    if (GC > G) GC = G;
    if (GC < 1) GC = 1;
    for (int g0 = 0; g0 < G; g0 += GC) {
      int gc = (G - g0 < GC) ? (G - g0) : GC;
      int M = gc * K;
      long long tot = (long long)M * cinPad;
      k_build_group_feat<<<(unsigned)((tot + 255) / 256), 256, 0, stream>>>(
          ptsH, Cp, xyzSrc, nxyz, ballIdx, K, g0, gc, N, S, PING, cinPad);
      g3(PING, L[0], PONG, L[0].cout, M, 1, 1);
      g3(PONG, L[1], PING, L[1].cout, M, 1, 1);
      g3(PING, L[2], PONG, L[2].cout, M, 1, 1);
      long long rtot = (long long)gc * L[2].cout;
      k_reduce_max<<<(unsigned)((rtot + 255) / 256), 256, 0, stream>>>(
          PONG, K, L[2].cout, g0, gc, outPts, ldOut, cOff, outF16);
    }
  };

  // ---- Stage 0: transpose xyz (B,3,N)->(B,N,3)
  {
    long long tot = (long long)B * N0 * 3;
    k_transpose_xyz<<<(unsigned)((tot + 255) / 256), 256, 0, stream>>>(
        (const float*)d_in[0], xyzT, B, N0);
  }

  // ---- SA1 (MSG): 4096 -> 512 points, 3 scales -> 320 channels
  k_fps<<<B, 256, 0, stream>>>(xyzT, N0, 512, fpsIdx1);
  k_gather3<<<(B * 512 * 3 + 255) / 256, 256, 0, stream>>>(xyzT, fpsIdx1, l1xyz, N0, 512, B);
  runScale(xyzT, N0, l1xyz, 512, nullptr, 0, 0.1f, 16,  sa1w[0], l1pts, 320, 0,   1);
  runScale(xyzT, N0, l1xyz, 512, nullptr, 0, 0.2f, 32,  sa1w[1], l1pts, 320, 64,  1);
  runScale(xyzT, N0, l1xyz, 512, nullptr, 0, 0.4f, 128, sa1w[2], l1pts, 320, 192, 1);

  // ---- SA2 (MSG): 512 -> 128 points, 3 scales -> 640 channels
  k_fps<<<B, 256, 0, stream>>>(l1xyz, 512, 128, fpsIdx2);
  k_gather3<<<(B * 128 * 3 + 255) / 256, 256, 0, stream>>>(l1xyz, fpsIdx2, l2xyz, 512, 128, B);
  runScale(l1xyz, 512, l2xyz, 128, l1pts, 320, 0.2f, 32,  sa2w[0], l2pts, 640, 0,   1);
  runScale(l1xyz, 512, l2xyz, 128, l1pts, 320, 0.4f, 64,  sa2w[1], l2pts, 640, 128, 1);
  runScale(l1xyz, 512, l2xyz, 128, l1pts, 320, 0.8f, 128, sa2w[2], l2pts, 640, 384, 1);

  // ---- SA3: per-point MLP 644 -> 256 -> 512 -> 1024 over 1024 points
  {
    int R = B * 128;
    long long tot = (long long)R * 672;
    k_build_feat3<<<(unsigned)((tot + 255) / 256), 256, 0, stream>>>(l2xyz, l2pts, PING, R, 672);
    g3(PING, sa3w[0], PONG, 256, R, 1, 1);
    g3(PONG, sa3w[1], PING, 512, R, 1, 1);
    g3(PING, sa3w[2], l3pts, 1024, R, 1, 1);
  }

  // ---- SA-app: 128 -> 32 points, one scale, 1027 -> 512 -> 256 -> 64
  k_fps<<<B, 256, 0, stream>>>(l2xyz, 128, 32, fpsIdx3);
  k_gather3<<<(B * 32 * 3 + 255) / 256, 256, 0, stream>>>(l2xyz, fpsIdx3, appxyz, 128, 32, B);
  runScale(l2xyz, 128, appxyz, 32, l3pts, 1024, 0.4f, 32, appw, memfeat, 64, 0, 0);

  // ---- Memory attention: q = l2norm(mem_feat); s = q @ l2norm(M)^T; addr=softmax;
  //      mf = addr @ M; then MLP 64 -> 128 -> 256
  k_l2norm_rows<<<4, 64, 0, stream>>>(memfeat, q_h, 256, 64, 64);
  k_l2norm_rows<<<1, 64, 0, stream>>>((const float*)d_in[53], memn_h, 16, 64, 64);
  k_memw_T<<<(64 * 32 + 255) / 256, 256, 0, stream>>>((const float*)d_in[53], memT_h);
  gemm(q_h,    memn_h, nullptr, s_f,  16, 256, 16, 64, 0, 0);
  k_softmax16<<<1, 256, 0, stream>>>(s_f, addr_h, 256);
  gemm(addr_h, memT_h, nullptr, mf_h, 64, 256, 64, 32, 0, 1);
  g3(mf_h,  memw[0], mm1_h, 128, 256, 1, 1);
  g3(mm1_h, memw[1], mm2_h, 256, 256, 1, 1);

  // ---- x = [max_p(mem mlp) (256) | max_p(l3_points) (1024)]
  k_reduce_max<<<(8 * 256 + 255) / 256, 256, 0, stream>>>(mm2_h, 32, 256, 0, 8, x_h, 1280, 0, 1);
  k_reduce_max<<<(8 * 1024 + 255) / 256, 256, 0, stream>>>(l3pts, 128, 1024, 0, 8, x_h, 1280, 256, 1);

  // ---- FC head + log_softmax
  g3(x_h,  fcw[0], t1_h, 512, B, 1, 1);
  g3(t1_h, fcw[1], t2_h, 256, B, 1, 1);
  g3(t2_h, fcw[2], logits, 40, B, 0, 0);
  k_logsoftmax<<<1, 32, 0, stream>>>(logits, (float*)d_out, 8, 40);
}
